// AFT2DAttention_89111981457894
// MI455X (gfx1250) — matrix operand order repaired and checked by reading, not compiled, not run on hardware
//
#include <hip/hip_runtime.h>
#include <hip/hip_bf16.h>

// ---------------------------------------------------------------------------
// Types for CDNA5 WMMA (wave32, 16x16x32 bf16 -> f32) and TDM descriptors
// ---------------------------------------------------------------------------
typedef __bf16 bf16_t;
typedef __attribute__((ext_vector_type(8)))  __bf16 v8bf;
typedef __attribute__((ext_vector_type(16))) __bf16 v16bf;
typedef __attribute__((ext_vector_type(8)))  float  v8f;
typedef __attribute__((ext_vector_type(4)))  unsigned int v4u;
typedef __attribute__((ext_vector_type(8)))  int v8i;
typedef __attribute__((ext_vector_type(4)))  int v4i;

__device__ __forceinline__ bf16_t f32_to_bf16(float f) {
  union { float f; unsigned u; } v; v.f = f;
  unsigned r = v.u + 0x7FFFu + ((v.u >> 16) & 1u);   // round-to-nearest-even
  union { unsigned short s; bf16_t b; } o; o.s = (unsigned short)(r >> 16);
  return o.b;
}

// ---------------------------------------------------------------------------
// TDM: async 2D tile load (rows x 256 bf16, K-contiguous) global -> LDS.
// Descriptor layout per CDNA5 ISA (D# groups 0/1).
// This toolchain exposes the 6-arg builtin: (g0, g1, g2, g3, g4, cpol).
// ---------------------------------------------------------------------------
__device__ __forceinline__ void tdm_load_rows_k256(unsigned ldsOff,
                                                   const bf16_t* gptr,
                                                   unsigned tensorRows,
                                                   unsigned tileRows) {
  unsigned long long ga = (unsigned long long)(const void*)gptr;
  v4u g0;
  g0[0] = 1u;                                          // count=1, load descriptor
  g0[1] = ldsOff;                                      // lds_addr (bytes)
  g0[2] = (unsigned)ga;                                // global_addr[31:0]
  g0[3] = (unsigned)((ga >> 32) & 0x1FFFFFFu)          // global_addr[56:32]
        | (2u << 30);                                  // type = 2 ("image")
  v8i g1;
  g1[0] = (int)(1u << 16);                             // data_size = 1 -> 2 bytes
  g1[1] = (int)(256u << 16);                           // tensor_dim0 = 256 (lo16)
  g1[2] = (int)((tensorRows & 0xFFFFu) << 16);         // dim0 hi=0 | tensor_dim1 lo16
  g1[3] = (int)((tensorRows >> 16) | (256u << 16));    // tensor_dim1 hi | tile_dim0=256
  g1[4] = (int)tileRows;                               // tile_dim1 | tile_dim2=0
  g1[5] = 256;                                         // tensor_dim0_stride[31:0]
  g1[6] = 0;                                           // stride0 hi | stride1 lo
  g1[7] = 0;                                           // stride1 hi
  v4i gz4 = {0, 0, 0, 0};                              // groups 2/3 unused (2D)
  v8i gz8 = {0, 0, 0, 0, 0, 0, 0, 0};                  // group 4 unused
  __builtin_amdgcn_tensor_load_to_lds(g0, g1, gz4, gz4, gz8, 0);
}

// ---------------------------------------------------------------------------
// fp32 -> bf16 conversion (4 elements / thread)
// ---------------------------------------------------------------------------
__global__ __launch_bounds__(256)
void cvt_f32_to_bf16(const float* __restrict__ in, bf16_t* __restrict__ out, int n4) {
  int i = blockIdx.x * blockDim.x + threadIdx.x;
  if (i >= n4) return;
  float4 f = ((const float4*)in)[i];
  union { bf16_t b[4]; unsigned long long u; } pk;
  pk.b[0] = f32_to_bf16(f.x);
  pk.b[1] = f32_to_bf16(f.y);
  pk.b[2] = f32_to_bf16(f.z);
  pk.b[3] = f32_to_bf16(f.w);
  ((unsigned long long*)out)[i] = pk.u;
}

// ---------------------------------------------------------------------------
// bf16 WMMA GEMM, TDM-staged:  out[row,col] = sum_k A[row,k] * Bw[col,k], K=256
// Block = 256 threads (8 waves). Tile: M=32, N=128.
// Whole A (32x256, 16KB) and B (128x256, 64KB) tiles land in LDS via two TDM
// descriptors (wave0/wave1), then 8 k-steps of pure ds_load + WMMA, barrier-free.
// Output columns [0,256) -> out0, [256,512) -> out1 (fused k/v GEMM).
// ---------------------------------------------------------------------------
#define GK 256
#define TM 32
#define TN 128
#define TK 32
#define LDS_A_BYTES (TM * GK * 2)                // 16384
#define LDS_B_BYTES (TN * GK * 2)                // 65536
#define GEMM_LDS    (LDS_A_BYTES + LDS_B_BYTES)  // 81920

__global__ __launch_bounds__(256)
void gemm_bf16_wmma(const bf16_t* __restrict__ A, const bf16_t* __restrict__ Bw,
                    float* __restrict__ out0, float* __restrict__ out1,
                    int bRows) {
  extern __shared__ __align__(16) char smem[];
  bf16_t* As = (bf16_t*)smem;                    // [TM][GK]
  bf16_t* Bs = (bf16_t*)(smem + LDS_A_BYTES);    // [TN][GK]

  const int t    = threadIdx.x;
  const int lane = t & 31;
  const int wave = t >> 5;       // 0..7
  const int wm   = wave & 1;     // M sub-tile (16 rows)
  const int wn   = wave >> 1;    // N sub-tile (32 cols)
  const long mBase = (long)blockIdx.x * TM;
  const long nBase = (long)blockIdx.y * TN;

  // --- async tile loads via Tensor Data Mover (one descriptor per tile) ---
  const unsigned ldsBase = (unsigned)(unsigned long long)(void*)smem;
  if (wave == 0) {
    tdm_load_rows_k256(ldsBase, A + mBase * GK, /*tensorRows*/ 0x7FFFFFFFu, TM);
  } else if (wave == 1) {
    tdm_load_rows_k256(ldsBase + LDS_A_BYTES, Bw + nBase * GK,
                       (unsigned)bRows - (unsigned)nBase, TN);
  }
  if (wave < 2) __builtin_amdgcn_s_wait_tensorcnt(0);
  __syncthreads();

  // --- 8 barrier-free k-steps of ds_load + WMMA ---
  v8f acc0 = {};
  v8f acc1 = {};
  const int am = wm * 16 + (lane & 15);
  const int kh = (lane >> 4) * 8;         // A: K sub-chunk per lane half
  const int kb = (lane >> 4) * 8;         // B: K sub-chunk per lane half

#pragma unroll
  for (int k0 = 0; k0 < GK; k0 += TK) {
    // A frag (ISA 7.12.2): lanes 0-15 hold K 0..7 & 16..23 of row M=lane.
    v8bf alo = *(const v8bf*)&As[am * GK + k0 + kh];
    v8bf ahi = *(const v8bf*)&As[am * GK + k0 + 16 + kh];
    v16bf afrag = __builtin_shufflevector(alo, ahi,
        0, 1, 2, 3, 4, 5, 6, 7, 8, 9, 10, 11, 12, 13, 14, 15);

#pragma unroll
    for (int nn = 0; nn < 2; ++nn) {
      const int bn = wn * 32 + nn * 16 + (lane & 15);
      v8bf blo = *(const v8bf*)&Bs[bn * GK + k0 + kb];
      v8bf bhi = *(const v8bf*)&Bs[bn * GK + k0 + kb + 16];
      v16bf bfrag = __builtin_shufflevector(blo, bhi,
          0, 1, 2, 3, 4, 5, 6, 7, 8, 9, 10, 11, 12, 13, 14, 15);
      if (nn == 0)
        acc0 = __builtin_amdgcn_wmma_f32_16x16x32_bf16(
            false, afrag, false, bfrag, (short)0, acc0, false, false);
      else
        acc1 = __builtin_amdgcn_wmma_f32_16x16x32_bf16(
            false, afrag, false, bfrag, (short)0, acc1, false, false);
    }
  }

  // C/D layout: lanes 0-15: (M=r, N=lane); lanes 16-31: (M=8+r, N=lane-16)
#pragma unroll
  for (int nn = 0; nn < 2; ++nn) {
    v8f acc = nn ? acc1 : acc0;
    const long col = nBase + wn * 32 + nn * 16 + (lane & 15);
    float* outp = (col < 256) ? out0 : out1;
    const long ccol = col & 255;
#pragma unroll
    for (int r = 0; r < 8; ++r) {
      const long row = mBase + wm * 16 + (lane >> 4) * 8 + r;
      outp[row * 256 + ccol] = acc[r];
    }
  }
}

// ---------------------------------------------------------------------------
// AFT 5x5 window stage. One block = 16x16 spatial tile of one batch image.
// LDS halo tile 20x20 x 16 channels of k (and v in pass 2). Out-of-image
// positions get k = -1e30 so exp() -> 0 implements the mask for free
// (this is why TDM zero-fill OOB cannot be used here).
// norm = sum over 25 offsets AND all 256 channels; o = sum(w*v)/(norm+1e-8).
// ---------------------------------------------------------------------------
#define RAD 2
#define WSZ 5
#define TILE 16
#define HALO (TILE + 2 * RAD)   // 20
#define CCH 16

__global__ __launch_bounds__(256)
void aft_window(const float* __restrict__ kbuf, const float* __restrict__ vbuf,
                const float* __restrict__ w_h, const float* __restrict__ w_v,
                bf16_t* __restrict__ o) {
  __shared__ float ks[HALO][HALO][CCH];   // 25.6 KB
  __shared__ float vs[HALO][HALO][CCH];   // 25.6 KB

  const int t    = threadIdx.x;
  const int b    = blockIdx.y;                 // batch 0..15
  const int tile = blockIdx.x;                 // 0..15
  const int th = (tile >> 2) * TILE;
  const int tw = (tile & 3) * TILE;
  const int ly = t >> 4, lx = t & 15;
  const int h = th + ly, w = tw + lx;

  float bias[WSZ][WSZ];
#pragma unroll
  for (int ii = 0; ii < WSZ; ++ii) {
    const float whv = w_h[ii];
#pragma unroll
    for (int jj = 0; jj < WSZ; ++jj)
      bias[ii][jj] = whv * (float)(ii - RAD - h) + w_v[jj] * (float)(jj - RAD - w);
  }

  const long imgBase = (long)b * 64 * 64 * 256;

  // ---- pass 1: global normalizer over window x channels ----
  float norm = 0.f;
  for (int ch = 0; ch < 256; ch += CCH) {
    for (int i = t; i < HALO * HALO * CCH; i += 256) {
      const int c  = i & (CCH - 1);
      const int ww = (i / CCH) % HALO;
      const int hh = i / (CCH * HALO);
      const int gh = th + hh - RAD, gw = tw + ww - RAD;
      float kv = -1e30f;
      if (gh >= 0 && gh < 64 && gw >= 0 && gw < 64)
        kv = kbuf[imgBase + ((long)gh * 64 + gw) * 256 + ch + c];
      ks[hh][ww][c] = kv;
    }
    __syncthreads();
#pragma unroll
    for (int ii = 0; ii < WSZ; ++ii)
#pragma unroll
      for (int jj = 0; jj < WSZ; ++jj) {
        const float bb = bias[ii][jj];
        const float* kp = &ks[ly + ii][lx + jj][0];
#pragma unroll
        for (int c = 0; c < CCH; ++c)
          norm += __expf(bb + kp[c]);
      }
    __syncthreads();
  }
  const float inv_norm = 1.f / (norm + 1e-8f);

  // ---- pass 2: weighted value aggregation, normalize, emit bf16 ----
  for (int ch = 0; ch < 256; ch += CCH) {
    for (int i = t; i < HALO * HALO * CCH; i += 256) {
      const int c  = i & (CCH - 1);
      const int ww = (i / CCH) % HALO;
      const int hh = i / (CCH * HALO);
      const int gh = th + hh - RAD, gw = tw + ww - RAD;
      float kv = -1e30f, vv = 0.f;
      if (gh >= 0 && gh < 64 && gw >= 0 && gw < 64) {
        const long idx = imgBase + ((long)gh * 64 + gw) * 256 + ch + c;
        kv = kbuf[idx];
        vv = vbuf[idx];
      }
      ks[hh][ww][c] = kv;
      vs[hh][ww][c] = vv;
    }
    __syncthreads();

    float acc[CCH];
#pragma unroll
    for (int c = 0; c < CCH; ++c) acc[c] = 0.f;
#pragma unroll
    for (int ii = 0; ii < WSZ; ++ii)
#pragma unroll
      for (int jj = 0; jj < WSZ; ++jj) {
        const float bb = bias[ii][jj];
        const float* kp = &ks[ly + ii][lx + jj][0];
        const float* vp = &vs[ly + ii][lx + jj][0];
#pragma unroll
        for (int c = 0; c < CCH; ++c) {
          const float wgt = __expf(bb + kp[c]);
          acc[c] += wgt * vp[c];
        }
      }

    const long outBase = imgBase + ((long)h * 64 + w) * 256 + ch;
#pragma unroll
    for (int c = 0; c < CCH; ++c)
      o[outBase + c] = f32_to_bf16(acc[c] * inv_norm);
    __syncthreads();
  }
}

// ---------------------------------------------------------------------------
// Orchestration
// ---------------------------------------------------------------------------
extern "C" void kernel_launch(void* const* d_in, const int* in_sizes, int n_in,
                              void* d_out, int out_size, void* d_ws, size_t ws_size,
                              hipStream_t stream) {
  (void)in_sizes; (void)n_in; (void)out_size; (void)ws_size;
  const float* x   = (const float*)d_in[0];   // (16,64,64,256)
  const float* whp = (const float*)d_in[1];   // (5,)
  const float* wvp = (const float*)d_in[2];   // (5,)
  const float* Wk  = (const float*)d_in[3];   // (256,256)
  const float* Wv  = (const float*)d_in[4];   // (256,256)
  const float* Wp  = (const float*)d_in[5];   // (256,256)
  float* out = (float*)d_out;

  const long NTOK = 16L * 64 * 64;            // 65536 tokens

  // Workspace layout (bytes):
  //   [0, 32M)      x_bf16  (reused as o_bf16 after GEMM1)
  //   [32M, +256K)  Wk||Wv bf16  (512 x 256)
  //   [.., +128K)   Wp bf16
  //   [.., +64M)    k fp32
  //   [.., +64M)    v fp32
  char* ws = (char*)d_ws;
  bf16_t* xb   = (bf16_t*)(ws);
  bf16_t* wkv  = (bf16_t*)(ws + 33554432);
  bf16_t* wpb  = (bf16_t*)(ws + 33816576);
  float*  kbuf = (float*) (ws + 33947648);
  float*  vbuf = (float*) (ws + 101056512);

  // 1) fp32 -> bf16 casts
  {
    int n4 = (int)(NTOK * 256 / 4);           // 4,194,304
    cvt_f32_to_bf16<<<(n4 + 255) / 256, 256, 0, stream>>>(x, xb, n4);
    cvt_f32_to_bf16<<<64, 256, 0, stream>>>(Wk, wkv,         65536 / 4);
    cvt_f32_to_bf16<<<64, 256, 0, stream>>>(Wv, wkv + 65536, 65536 / 4);
    cvt_f32_to_bf16<<<64, 256, 0, stream>>>(Wp, wpb,         65536 / 4);
  }

  // 2) fused k/v GEMM: [65536 x 256] x [512 x 256]^T -> k, v
  {
    dim3 grid((unsigned)(NTOK / TM), 512 / TN);   // (2048, 4)
    gemm_bf16_wmma<<<grid, 256, GEMM_LDS, stream>>>(xb, wkv, kbuf, vbuf, 512);
  }

  // 3) AFT 5x5 window aggregation -> o (bf16, overwrites x_bf16 region)
  {
    dim3 grid(16, 16);                            // 16 tiles x 16 batches
    aft_window<<<grid, 256, 0, stream>>>(kbuf, vbuf, whp, wvp, xb);
  }

  // 4) projection GEMM: o x Wp^T -> out (fp32)
  {
    dim3 grid((unsigned)(NTOK / TM), 256 / TN);   // (2048, 2)
    gemm_bf16_wmma<<<grid, 256, GEMM_LDS, stream>>>(xb, wpb, out, out, 256);
  }
}
